// LogitsNew_41566693490992
// MI455X (gfx1250) — matrix operand order repaired and checked by reading, not compiled, not run on hardware
//
#include <hip/hip_runtime.h>
#include <math.h>

// Problem constants (from reference): B=32, N=256, D=C=1024
#define D_DIM 1024
#define C_DIM 1024
#define M_DIM 8192   // B*N

typedef __attribute__((ext_vector_type(2)))  float  v2f;
typedef __attribute__((ext_vector_type(8)))  float  v8f;
typedef __attribute__((ext_vector_type(16))) __bf16 v16bf;

__device__ __forceinline__ v8f wmma_f32_16x16x4(v2f a, v2f b, v8f c) {
  return __builtin_amdgcn_wmma_f32_16x16x4_f32(false, a, false, b, (short)0, c,
                                               false, false);
}
__device__ __forceinline__ v8f wmma_bf16(v16bf a, v16bf b, v8f c) {
  return __builtin_amdgcn_wmma_f32_16x16x32_bf16(false, a, false, b, (short)0,
                                                 c, false, false);
}

// ---- bf16 split helpers (x = hi + lo, both bf16, RNE) ----------------------
__device__ __forceinline__ float bfval(unsigned short b) {
  unsigned int u = (unsigned int)b << 16;
  return __builtin_bit_cast(float, u);
}
// Pack split of (x0, x1): hi = {bf16(x1),bf16(x0)}, lo = residuals, k0 in LSBs.
__device__ __forceinline__ void split2(float x0, float x1, unsigned int& hi,
                                       unsigned int& lo) {
  unsigned short h0 = __builtin_bit_cast(unsigned short, (__bf16)x0);
  unsigned short h1 = __builtin_bit_cast(unsigned short, (__bf16)x1);
  hi = (unsigned int)h0 | ((unsigned int)h1 << 16);
  float r0 = x0 - bfval(h0);
  float r1 = x1 - bfval(h1);
  unsigned short l0 = __builtin_bit_cast(unsigned short, (__bf16)r0);
  unsigned short l1 = __builtin_bit_cast(unsigned short, (__bf16)r1);
  lo = (unsigned int)l0 | ((unsigned int)l1 << 16);
}

// ---------------------------------------------------------------------------
// Small GEMM: (32 x 1024) @ (1024 x 1024) -> (32 x 1024), optional bias+GELU.
// One wave per 16x16 tile, exact-f32 WMMA over K. Everything L2-resident.
// ---------------------------------------------------------------------------
__global__ __launch_bounds__(256) void small_gemm_kernel(
    const float* __restrict__ A, const float* __restrict__ Bm,
    const float* __restrict__ bias, float* __restrict__ out, int apply_gelu) {
  const int tid = threadIdx.x;
  const int ln  = tid & 31;
  const int w   = tid >> 5;
  const int wg  = blockIdx.x * 8 + w;   // 128 tiles total
  const int m0  = (wg & 1) * 16;
  const int n0  = (wg >> 1) * 16;
  const int mi  = ln & 15;
  const int kh  = ln >> 4;

  v8f acc = {};
  const float* aptr = A + (m0 + mi) * D_DIM + 2 * kh;
  const float* bptr = Bm + (2 * kh) * C_DIM + (n0 + mi);

#pragma unroll 4
  for (int k = 0; k < D_DIM; k += 4) {
    v2f a = *(const v2f*)(aptr + k);
    v2f b;
    b.x = bptr[k * C_DIM];
    b.y = bptr[k * C_DIM + C_DIM];
    acc = wmma_f32_16x16x4(a, b, acc);
  }

  const float bv = bias ? bias[n0 + mi] : 0.0f;
  const int row0 = m0 + 8 * kh;
#pragma unroll
  for (int r = 0; r < 8; ++r) {
    float x = acc[r] + bv;
    if (apply_gelu)
      x = 0.5f * x * (1.0f + erff(x * 0.70710678118654752f));  // exact GELU
    out[(row0 + r) * C_DIM + n0 + mi] = x;
  }
}

// ---------------------------------------------------------------------------
// Main GEMM (bf16x3): (8192x1024) @ (1024x1024) -> (8192x1024) + broadcast v.
// 256 threads = 8 waves; block tile 128x64, BK=32; double-buffered LDS with
// on-the-fly f32 -> (bf16 hi, bf16 lo) split.  Each K=32 slab = 3 bf16 WMMAs
// per subtile instead of 8 f32 WMMAs (~2.7x fewer matrix instructions).
// ---------------------------------------------------------------------------
#define BM 128
#define BN 64
#define BK 32
#define APAD 20   // A row stride in dwords (= 40 bf16, perfect 2-way banks)
#define BPAD 20   // B^T row stride in dwords

__global__ __launch_bounds__(256) void main_gemm_bf16x3(
    const float* __restrict__ EE,   // M x D row-major
    const float* __restrict__ We,   // D x C row-major
    const float* __restrict__ V,    // 32 x C
    float* __restrict__ Out) {      // M x C
  __shared__ unsigned int AsHi[2][BM * APAD];   // [m][k] bf16 pairs
  __shared__ unsigned int AsLo[2][BM * APAD];
  __shared__ unsigned int BtHi[2][BN * BPAD];   // [n][k] bf16 pairs (transposed)
  __shared__ unsigned int BtLo[2][BN * BPAD];

  const int tid = threadIdx.x;
  const int ln  = tid & 31;
  const int w   = tid >> 5;
  const int blockM = blockIdx.y * BM;
  const int blockN = blockIdx.x * BN;
  const int wm = (w & 3) * 32;
  const int wn = (w >> 2) * 32;
  const int mi = ln & 15;
  const int kh = ln >> 4;

  // Global->reg staging: A = 128x32 f32 (16 f32/thread), B = 32x64 f32 (8).
  const int arow  = tid >> 3;          // + 32*s, s=0..3
  const int akoff = (tid & 7) * 4;     // f32 offset within K-chunk
  const int bkp   = tid >> 4;          // k-pair index 0..15
  const int bcol  = (tid & 15) * 4;    // 4 consecutive columns

  const float* Ag = EE + blockM * D_DIM;
  const float* Bg = We + blockN;

  float4 ra[4], rb0, rb1;
  auto gload = [&](int k0) {
#pragma unroll
    for (int s = 0; s < 4; ++s)
      ra[s] = *(const float4*)(Ag + (arow + 32 * s) * D_DIM + k0 + akoff);
    rb0 = *(const float4*)(Bg + (k0 + 2 * bkp) * C_DIM + bcol);
    rb1 = *(const float4*)(Bg + (k0 + 2 * bkp + 1) * C_DIM + bcol);
  };
  auto sstore = [&](int buf) {
#pragma unroll
    for (int s = 0; s < 4; ++s) {
      unsigned int h0, l0, h1, l1;
      split2(ra[s].x, ra[s].y, h0, l0);
      split2(ra[s].z, ra[s].w, h1, l1);
      const int idx = (arow + 32 * s) * APAD + (akoff >> 1);  // even -> b64
      uint2 ph; ph.x = h0; ph.y = h1;
      uint2 pl; pl.x = l0; pl.y = l1;
      *(uint2*)&AsHi[buf][idx] = ph;
      *(uint2*)&AsLo[buf][idx] = pl;
    }
#pragma unroll
    for (int i = 0; i < 4; ++i) {
      unsigned int h, l;
      split2((&rb0.x)[i], (&rb1.x)[i], h, l);   // (k, k+1) pair at column
      BtHi[buf][(bcol + i) * BPAD + bkp] = h;
      BtLo[buf][(bcol + i) * BPAD + bkp] = l;
    }
  };

  // Fragment loads per ISA 16-bit layouts (wave32):
  // A 16x32: lane(m,kh) needs K in [8kh,8kh+8) U [16+8kh,16+8kh+8).
  auto ldfragA = [&](const unsigned int* base, int row) -> v16bf {
    struct { uint4 a, b; } r;
    const unsigned int* p = base + row * APAD + 4 * kh;
    r.a = *(const uint4*)(p);        // ds_load_b128
    r.b = *(const uint4*)(p + 8);
    return __builtin_bit_cast(v16bf, r);
  };
  // B 32x16: lane(n,kh) needs K in [8kh,8kh+8) U [16+8kh,16+8kh+8).
  auto ldfragB = [&](const unsigned int* base, int row) -> v16bf {
    struct { uint4 a, b; } r;
    const unsigned int* p = base + row * BPAD + 4 * kh;
    r.a = *(const uint4*)(p);
    r.b = *(const uint4*)(p + 8);
    return __builtin_bit_cast(v16bf, r);
  };

  v8f acc[2][2] = {};

  gload(0);
  sstore(0);
  gload(BK);
  __syncthreads();

  const int NKT = D_DIM / BK;   // 32
  for (int kt = 0; kt < NKT; ++kt) {
    const int cur = kt & 1;
    if (kt + 1 < NKT) sstore(1 - cur);           // regs hold tile kt+1
    if (kt + 2 < NKT) gload((kt + 2) * BK);      // prefetch 2 ahead

    const unsigned int* AH = &AsHi[cur][0];
    const unsigned int* AL = &AsLo[cur][0];
    const unsigned int* BH = &BtHi[cur][0];
    const unsigned int* BL = &BtLo[cur][0];

    v16bf ah[2], al[2];
    ah[0] = ldfragA(AH, wm + mi);
    ah[1] = ldfragA(AH, wm + 16 + mi);
    al[0] = ldfragA(AL, wm + mi);
    al[1] = ldfragA(AL, wm + 16 + mi);
#pragma unroll
    for (int j = 0; j < 2; ++j) {
      v16bf bh = ldfragB(BH, wn + 16 * j + mi);
      v16bf bl = ldfragB(BL, wn + 16 * j + mi);
#pragma unroll
      for (int i = 0; i < 2; ++i) {
        acc[i][j] = wmma_bf16(ah[i], bh, acc[i][j]);  // hi*hi
        acc[i][j] = wmma_bf16(ah[i], bl, acc[i][j]);  // hi*lo
        acc[i][j] = wmma_bf16(al[i], bh, acc[i][j]);  // lo*hi
      }
    }
    __syncthreads();
  }

  // Epilogue: broadcast v[b,:] (each 128-row block lies in a single b).
  const float* vrow = V + (blockM >> 8) * C_DIM;
#pragma unroll
  for (int j = 0; j < 2; ++j) {
    const int col = blockN + wn + 16 * j + mi;
    const float vadd = vrow[col];
#pragma unroll
    for (int i = 0; i < 2; ++i) {
      const int row0 = blockM + wm + 16 * i + 8 * kh;
#pragma unroll
      for (int r = 0; r < 8; ++r) {
        Out[(row0 + r) * C_DIM + col] = acc[i][j][r] + vadd;
      }
    }
  }
}

// ---------------------------------------------------------------------------
extern "C" void kernel_launch(void* const* d_in, const int* in_sizes, int n_in,
                              void* d_out, int out_size, void* d_ws,
                              size_t ws_size, hipStream_t stream) {
  (void)in_sizes; (void)n_in; (void)out_size; (void)ws_size;
  const float* E  = (const float*)d_in[0];  // (32, 1024)
  const float* EE = (const float*)d_in[1];  // (32, 256, 1024)
  const float* Wm = (const float*)d_in[2];  // (2048, 1024)
  const float* Wp = (const float*)d_in[3];  // (1024, 1024)
  const float* bp = (const float*)d_in[4];  // (1024,)
  float* out = (float*)d_out;               // (32, 256, 1024)

  float* u = (float*)d_ws;                  // 32*1024 f32
  float* v = u + 32 * C_DIM;                // 32*1024 f32

  const float* Wu = Wm;                     // rows [0, 1024)
  const float* We = Wm + D_DIM * C_DIM;     // rows [1024, 2048)

  // u = gelu(E @ W_proj + b)
  small_gemm_kernel<<<16, 256, 0, stream>>>(E, Wp, bp, u, 1);
  // v = u @ W_u
  small_gemm_kernel<<<16, 256, 0, stream>>>(u, Wu, nullptr, v, 0);
  // out = EE @ W_e + v[:, None, :]   (bf16x3 split GEMM)
  dim3 grid(C_DIM / BN, M_DIM / BM);        // (16, 64)
  main_gemm_bf16x3<<<grid, 256, 0, stream>>>(EE, We, v, out);
}
